// FusionOnManifold_71476845740057
// MI455X (gfx1250) — matrix operand order repaired and checked by reading, not compiled, not run on hardware
//
#include <hip/hip_runtime.h>
#include <math.h>

// ---------------------------------------------------------------------------
// Types for CDNA5 WMMA (wave32). v16bf = 8 VGPRs of bf16 pairs.
// ---------------------------------------------------------------------------
typedef __attribute__((ext_vector_type(16))) __bf16        v16bf;
typedef __attribute__((ext_vector_type(8)))  float         v8f;
typedef __attribute__((ext_vector_type(4)))  unsigned int  v4u;
typedef __attribute__((ext_vector_type(4)))  int           v4i;
typedef __attribute__((ext_vector_type(4)))  float         v4f;

union Frag {
    v16bf v;
    v4u q[2];
    unsigned short h[16];
};

#if defined(__HIP_DEVICE_COMPILE__) && \
    __has_builtin(__builtin_amdgcn_global_load_async_to_lds_b128)
#define HAVE_ASYNC_LDS 1
#else
#define HAVE_ASYNC_LDS 0
#endif

#define AS_GLOBAL __attribute__((address_space(1)))
#define AS_LOCAL  __attribute__((address_space(3)))

__device__ __forceinline__ unsigned short f2bf(float f) {
    unsigned int u = __float_as_uint(f);
    unsigned int r = u + 0x7FFFu + ((u >> 16) & 1u);  // round-to-nearest-even
    return (unsigned short)(r >> 16);
}

__device__ __forceinline__ float gelu_exact(float x) {
    return 0.5f * x * (1.0f + erff(x * 0.70710678118654752f));
}

// Stage one 16-byte chunk of the B tile into LDS (async DMA when available).
__device__ __forceinline__ void stage16(const unsigned short* g, unsigned short* l) {
#if HAVE_ASYNC_LDS
    __builtin_amdgcn_global_load_async_to_lds_b128(
        (AS_GLOBAL v4i*)(g), (AS_LOCAL v4i*)(l), 0, 0);
#else
    *(v4u*)l = *(const v4u*)g;
#endif
}

__device__ __forceinline__ void wait_async_stage() {
#if HAVE_ASYNC_LDS
#if __has_builtin(__builtin_amdgcn_s_wait_asynccnt)
    __builtin_amdgcn_s_wait_asynccnt(0);
#else
    asm volatile("s_wait_asynccnt 0x0" ::: "memory");
#endif
#endif
}

// ---------------------------------------------------------------------------
// Transpose + convert f32 -> bf16 with zero padding.
// dst[n*dstK + k] = (n < srcN && k < srcK) ? bf16(src[k*srcN + n]) : 0
// ---------------------------------------------------------------------------
__global__ void k_transpose_bf16(const float* __restrict__ src,
                                 unsigned short* __restrict__ dst,
                                 int srcK, int srcN, int dstRows, int dstK) {
    long total = (long)dstRows * dstK;
    for (long t = blockIdx.x * (long)blockDim.x + threadIdx.x; t < total;
         t += (long)gridDim.x * blockDim.x) {
        int n = (int)(t / dstK);
        int k = (int)(t % dstK);
        unsigned short v = 0;
        if (n < srcN && k < srcK) v = f2bf(src[(long)k * srcN + n]);
        dst[t] = v;
    }
}

// ---------------------------------------------------------------------------
// Block reduction (256 threads = 8 wave32), broadcast result to all threads.
// ---------------------------------------------------------------------------
__device__ __forceinline__ float blockReduceSum(float val, float* s8) {
    #pragma unroll
    for (int off = 16; off > 0; off >>= 1) val += __shfl_down(val, off);
    int lane = threadIdx.x & 31, w = threadIdx.x >> 5;
    if (lane == 0) s8[w] = val;
    __syncthreads();
    float r = (threadIdx.x < 8) ? s8[threadIdx.x] : 0.0f;
    if (w == 0) {
        #pragma unroll
        for (int off = 4; off > 0; off >>= 1) r += __shfl_down(r, off);
        if (lane == 0) s8[0] = r;
    }
    __syncthreads();
    float out = s8[0];
    __syncthreads();
    return out;
}

#define MINK_SIGN(i) ((i) == 0 ? -1.0f : 1.0f)
#define HEPS 1e-6f

// ---------------------------------------------------------------------------
// y_tan = log_map(anchor, y_h) per row; write bf16, padded to APAD cols.
// ---------------------------------------------------------------------------
__global__ __launch_bounds__(256)
void k_logmap_rows(const float* __restrict__ y_h, const float* __restrict__ anchor,
                   unsigned short* __restrict__ y_tan, int A, int APAD) {
    __shared__ float s8[8];
    long t = blockIdx.x;
    const float* y = y_h + t * A;
    int i0 = threadIdx.x, i1 = threadIdx.x + 256;
    float p[2] = {0, 0}, yv[2] = {0, 0};
    int cnt = (i1 < A) ? 2 : 1;
    p[0] = anchor[i0]; yv[0] = y[i0];
    if (cnt == 2) { p[1] = anchor[i1]; yv[1] = y[i1]; }

    float part = MINK_SIGN(i0) * p[0] * yv[0];
    if (cnt == 2) part += p[1] * yv[1];
    float a = fmaxf(-blockReduceSum(part, s8), 1.0f + HEPS);

    float u[2]; u[0] = yv[0] - a * p[0]; u[1] = yv[1] - a * p[1];
    part = MINK_SIGN(i0) * u[0] * u[0];
    if (cnt == 2) part += u[1] * u[1];
    float nu = sqrtf(fmaxf(blockReduceSum(part, s8), HEPS));
    float s = acoshf(a) / nu;

    y_tan[t * APAD + i0] = f2bf(s * u[0]);
    if (cnt == 2) y_tan[t * APAD + i1] = f2bf(s * u[1]);
    for (int i = A + threadIdx.x; i < APAD; i += 256) y_tan[t * APAD + i] = 0;
}

// ---------------------------------------------------------------------------
// Generic bf16 WMMA GEMM: C[M,N] = A[M,K] @ Bt[N,K]^T + bias.
// Block = 8 waves, tile 128 rows x 64 cols; wave tile 16x64 (4 accumulators).
// B tile (64 x 32 bf16) staged in LDS (double buffered, async DMA when
// available) and shared across all 8 waves. A may be f32 (converted on the
// fly) or bf16. Output f32 or bf16.
// ---------------------------------------------------------------------------
#define BROWS 40  // LDS row pitch in bf16 (80 B): conflict-free ds_load_b128

template <bool A_IS_F32, bool OUT_BF16>
__global__ __launch_bounds__(256)
void k_gemm_wmma(const void* __restrict__ Aptr, const unsigned short* __restrict__ Bt,
                 const float* __restrict__ bias, int nbias, void* __restrict__ Cptr,
                 int K, int lda, int ldb, int ldc) {
    __shared__ unsigned short ldsB[2][64 * BROWS];

    const int tid = threadIdx.x;
    const int lane = tid & 31;
    const int wave = tid >> 5;
    const int m16 = lane & 15;
    const int khalf = lane >> 4;
    const long row0 = (long)blockIdx.y * 128 + wave * 16;
    const int col0 = blockIdx.x * 64;
    const long am = row0 + m16;

    // This thread's 16-byte staging chunk of the 64x32 B tile.
    const int sn = tid >> 2;        // row in tile (0..63)
    const int sk = (tid & 3) * 8;   // bf16 offset within row (0,8,16,24)
    const unsigned short* gB = Bt + (long)(col0 + sn) * ldb + sk;
    unsigned short* lB[2] = {&ldsB[0][sn * BROWS + sk], &ldsB[1][sn * BROWS + sk]};

    v8f acc[4];
    v8f zero = {0, 0, 0, 0, 0, 0, 0, 0};
    #pragma unroll
    for (int c = 0; c < 4; ++c) acc[c] = zero;

    // Prologue: fill buffer 0 with the k=0 tile.
    stage16(gB, lB[0]);
    wait_async_stage();
    __syncthreads();

    for (int kk = 0; kk < K; kk += 32) {
        const int buf = (kk >> 5) & 1;
        // Prefetch next k-step's B tile into the other buffer.
        if (kk + 32 < K) stage16(gB + kk + 32, lB[buf ^ 1]);

        Frag a;
        if constexpr (A_IS_F32) {
            const float* ap = (const float*)Aptr + am * lda + kk + 8 * khalf;
            v4f f0 = *(const v4f*)(ap);
            v4f f1 = *(const v4f*)(ap + 4);
            v4f f2 = *(const v4f*)(ap + 16);
            v4f f3 = *(const v4f*)(ap + 20);
            #pragma unroll
            for (int j = 0; j < 4; ++j) {
                a.h[j]      = f2bf(f0[j]);
                a.h[4 + j]  = f2bf(f1[j]);
                a.h[8 + j]  = f2bf(f2[j]);
                a.h[12 + j] = f2bf(f3[j]);
            }
        } else {
            const unsigned short* ap =
                (const unsigned short*)Aptr + am * lda + kk + 8 * khalf;
            a.q[0] = *(const v4u*)(ap);
            a.q[1] = *(const v4u*)(ap + 16);
        }

        const unsigned short* lbase = &ldsB[buf][0];
        #pragma unroll
        for (int c = 0; c < 4; ++c) {
            const unsigned short* bp = lbase + (16 * c + m16) * BROWS + khalf * 8;
            Frag b;
            b.q[0] = *(const v4u*)(bp);
            b.q[1] = *(const v4u*)(bp + 16);
            acc[c] = __builtin_amdgcn_wmma_f32_16x16x32_bf16(
                false, a.v, false, b.v, (short)0, acc[c], false, false);
        }

        wait_async_stage();  // own async fill of next buffer complete
        __syncthreads();     // everyone's fill visible / reads of buf done
    }

    #pragma unroll
    for (int c = 0; c < 4; ++c) {
        int n = col0 + 16 * c + m16;
        float bv = (n < nbias) ? bias[n] : 0.0f;
        #pragma unroll
        for (int v = 0; v < 8; ++v) {
            long m = row0 + v + 8 * khalf;
            float val = acc[c][v] + bv;
            if constexpr (OUT_BF16)
                ((unsigned short*)Cptr)[m * ldc + n] = f2bf(val);
            else
                ((float*)Cptr)[m * ldc + n] = val;
        }
    }
}

// ---------------------------------------------------------------------------
// Gate: gate_h = gelu([e_out | h_feat] @ W_g1 + b_g1); alpha = sigmoid(gate_h@W_g2+b_g2)
// K = 4096 (first 2048 from f32 e_out, rest from bf16 h_feat). N = 64 (4 tiles).
// ---------------------------------------------------------------------------
__global__ __launch_bounds__(256)
void k_gate_alpha(const float* __restrict__ e_out, const unsigned short* __restrict__ h_feat,
                  const unsigned short* __restrict__ Wt_g1, const float* __restrict__ b_g1,
                  const float* __restrict__ W_g2, const float* __restrict__ b_g2,
                  float* __restrict__ alpha, int D) {
    const int lane = threadIdx.x & 31;
    const int wave = threadIdx.x >> 5;
    const int m16 = lane & 15;
    const int khalf = lane >> 4;
    const long row0 = (long)blockIdx.x * 128 + wave * 16;
    const long am = row0 + m16;
    const int K = 2 * D;

    v8f acc[4];
    v8f zero = {0, 0, 0, 0, 0, 0, 0, 0};
    #pragma unroll
    for (int c = 0; c < 4; ++c) acc[c] = zero;

    for (int kk = 0; kk < K; kk += 32) {
        Frag a;
        if (kk < D) {
            const float* ap = e_out + am * D + kk + 8 * khalf;
            v4f f0 = *(const v4f*)(ap);
            v4f f1 = *(const v4f*)(ap + 4);
            v4f f2 = *(const v4f*)(ap + 16);
            v4f f3 = *(const v4f*)(ap + 20);
            #pragma unroll
            for (int j = 0; j < 4; ++j) {
                a.h[j]      = f2bf(f0[j]);
                a.h[4 + j]  = f2bf(f1[j]);
                a.h[8 + j]  = f2bf(f2[j]);
                a.h[12 + j] = f2bf(f3[j]);
            }
        } else {
            const unsigned short* ap = h_feat + am * D + (kk - D) + 8 * khalf;
            a.q[0] = *(const v4u*)(ap);
            a.q[1] = *(const v4u*)(ap + 16);
        }
        #pragma unroll
        for (int c = 0; c < 4; ++c) {
            const unsigned short* bp = Wt_g1 + (long)(16 * c + m16) * K + kk + 8 * khalf;
            Frag b;
            b.q[0] = *(const v4u*)(bp);
            b.q[1] = *(const v4u*)(bp + 16);
            acc[c] = __builtin_amdgcn_wmma_f32_16x16x32_bf16(
                false, a.v, false, b.v, (short)0, acc[c], false, false);
        }
    }

    float bg[4], wg[4];
    #pragma unroll
    for (int c = 0; c < 4; ++c) {
        int n = 16 * c + m16;
        bg[c] = b_g1[n];
        wg[c] = W_g2[n];
    }
    float b2 = b_g2[0];
    #pragma unroll
    for (int v = 0; v < 8; ++v) {
        float s = 0.0f;
        #pragma unroll
        for (int c = 0; c < 4; ++c) s += gelu_exact(acc[c][v] + bg[c]) * wg[c];
        s += __shfl_xor(s, 1);
        s += __shfl_xor(s, 2);
        s += __shfl_xor(s, 4);
        s += __shfl_xor(s, 8);
        if (m16 == 0) {
            long m = row0 + v + 8 * khalf;
            alpha[m] = 1.0f / (1.0f + expf(-(s + b2)));
        }
    }
}

// ---------------------------------------------------------------------------
// Per-row manifold chain:
//   v_tan = project_tangent(p, v_raw); x_e = exp_map(p, v_tan);
//   z_m = exp_map(x_e, alpha * log_map(x_e, y_h)); z_tan = log_map(p, z_m)
// ---------------------------------------------------------------------------
__global__ __launch_bounds__(256)
void k_manifold(const float* __restrict__ y_h, const float* __restrict__ anchor,
                const float* __restrict__ v_raw, const float* __restrict__ alpha,
                unsigned short* __restrict__ z_tan, int A, int APAD, int VPAD) {
    __shared__ float s8[8];
    long t = blockIdx.x;
    const float* y = y_h + t * A;
    const float* vr = v_raw + t * VPAD;
    float al = alpha[t];
    int i0 = threadIdx.x, i1 = threadIdx.x + 256;
    int cnt = (i1 < A) ? 2 : 1;

    float p[2] = {0, 0}, yv[2] = {0, 0}, vrv[2] = {0, 0};
    p[0] = anchor[i0]; yv[0] = y[i0]; vrv[0] = vr[i0];
    if (cnt == 2) { p[1] = anchor[i1]; yv[1] = y[i1]; vrv[1] = vr[i1]; }
    float sg0 = MINK_SIGN(i0);

    // project_tangent: v_tan = v_raw + <p, v_raw>_L * p
    float part = sg0 * p[0] * vrv[0];
    if (cnt == 2) part += p[1] * vrv[1];
    float ip = blockReduceSum(part, s8);
    float vt[2];
    vt[0] = vrv[0] + ip * p[0];
    vt[1] = vrv[1] + ip * p[1];

    // x_e = exp_map(p, v_tan)
    part = sg0 * vt[0] * vt[0];
    if (cnt == 2) part += vt[1] * vt[1];
    float nv = sqrtf(fmaxf(blockReduceSum(part, s8), HEPS));
    float ch = coshf(nv), sh = sinhf(nv) / nv;
    float xe[2];
    xe[0] = ch * p[0] + sh * vt[0];
    xe[1] = ch * p[1] + sh * vt[1];

    // l = log_map(x_e, y_h)
    part = sg0 * xe[0] * yv[0];
    if (cnt == 2) part += xe[1] * yv[1];
    float a1 = fmaxf(-blockReduceSum(part, s8), 1.0f + HEPS);
    float u[2];
    u[0] = yv[0] - a1 * xe[0];
    u[1] = yv[1] - a1 * xe[1];
    part = sg0 * u[0] * u[0];
    if (cnt == 2) part += u[1] * u[1];
    float nu = sqrtf(fmaxf(blockReduceSum(part, s8), HEPS));
    float s1 = acoshf(a1) / nu;

    // w = alpha * l ; z_m = exp_map(x_e, w)
    float w[2];
    w[0] = al * s1 * u[0];
    w[1] = al * s1 * u[1];
    part = sg0 * w[0] * w[0];
    if (cnt == 2) part += w[1] * w[1];
    float nw = sqrtf(fmaxf(blockReduceSum(part, s8), HEPS));
    float chw = coshf(nw), shw = sinhf(nw) / nw;
    float zm[2];
    zm[0] = chw * xe[0] + shw * w[0];
    zm[1] = chw * xe[1] + shw * w[1];

    // z_tan = log_map(p, z_m)
    part = sg0 * p[0] * zm[0];
    if (cnt == 2) part += p[1] * zm[1];
    float a2 = fmaxf(-blockReduceSum(part, s8), 1.0f + HEPS);
    float u2[2];
    u2[0] = zm[0] - a2 * p[0];
    u2[1] = zm[1] - a2 * p[1];
    part = sg0 * u2[0] * u2[0];
    if (cnt == 2) part += u2[1] * u2[1];
    float nu2 = sqrtf(fmaxf(blockReduceSum(part, s8), HEPS));
    float s2 = acoshf(a2) / nu2;

    z_tan[t * APAD + i0] = f2bf(s2 * u2[0]);
    if (cnt == 2) z_tan[t * APAD + i1] = f2bf(s2 * u2[1]);
    for (int i = A + threadIdx.x; i < APAD; i += 256) z_tan[t * APAD + i] = 0;
}

// ---------------------------------------------------------------------------
// Launcher
// ---------------------------------------------------------------------------
extern "C" void kernel_launch(void* const* d_in, const int* in_sizes, int n_in,
                              void* d_out, int out_size, void* d_ws, size_t ws_size,
                              hipStream_t stream) {
    const float* e_out  = (const float*)d_in[0];
    const float* y_h    = (const float*)d_in[1];
    const float* anchor = (const float*)d_in[2];
    const float* W_hlog = (const float*)d_in[3];
    const float* b_hlog = (const float*)d_in[4];
    const float* W_e2a  = (const float*)d_in[5];
    const float* b_e2a  = (const float*)d_in[6];
    const float* W_out  = (const float*)d_in[7];
    const float* b_out  = (const float*)d_in[8];
    const float* W_g1   = (const float*)d_in[9];
    const float* b_g1   = (const float*)d_in[10];
    const float* W_g2   = (const float*)d_in[11];
    const float* b_g2   = (const float*)d_in[12];

    const long NT = 16384;           // B*T
    const int D = 2048, A = 257, GH = 64;
    const int APAD = 288;            // 257 -> 9 * 32 (WMMA K multiple)
    const int VPAD = 320;            // 257 -> 5 * 64 (col-tile multiple)
    const int K2 = 2 * D;            // 4096

    char* ws = (char*)d_ws;
    unsigned short* y_tan  = (unsigned short*)ws; ws += (size_t)NT * APAD * 2;
    unsigned short* z_tan  = (unsigned short*)ws; ws += (size_t)NT * APAD * 2;
    unsigned short* h_feat = (unsigned short*)ws; ws += (size_t)NT * D * 2;
    float*          v_raw  = (float*)ws;          ws += (size_t)NT * VPAD * 4;
    unsigned short* Wt_hlog = (unsigned short*)ws; ws += (size_t)D * APAD * 2;
    unsigned short* Wt_e2a  = (unsigned short*)ws; ws += (size_t)VPAD * D * 2;
    unsigned short* Wt_g1   = (unsigned short*)ws; ws += (size_t)GH * K2 * 2;
    unsigned short* Wt_out  = (unsigned short*)ws; ws += (size_t)D * APAD * 2;

    float* z_fused = (float*)d_out;
    float* alpha   = (float*)d_out + (size_t)NT * D;

    // 1) weight transpose+convert to bf16 (K-padded, zero-filled)
    k_transpose_bf16<<<1024, 256, 0, stream>>>(W_hlog, Wt_hlog, A, D, D, APAD);
    k_transpose_bf16<<<1024, 256, 0, stream>>>(W_e2a, Wt_e2a, D, A, VPAD, D);
    k_transpose_bf16<<<1024, 256, 0, stream>>>(W_g1, Wt_g1, K2, GH, GH, K2);
    k_transpose_bf16<<<1024, 256, 0, stream>>>(W_out, Wt_out, A, D, D, APAD);

    // 2) y_tan = log_map(anchor, y_h)
    k_logmap_rows<<<NT, 256, 0, stream>>>(y_h, anchor, y_tan, A, APAD);

    // 3) h_feat = y_tan @ W_hlog + b_hlog  (bf16 out)
    k_gemm_wmma<false, true><<<dim3(D / 64, NT / 128), 256, 0, stream>>>(
        y_tan, Wt_hlog, b_hlog, D, h_feat, APAD, APAD, APAD, D);

    // 4) v_raw = e_out @ W_e2a + b_e2a  (f32 out, padded cols)
    k_gemm_wmma<true, false><<<dim3(VPAD / 64, NT / 128), 256, 0, stream>>>(
        e_out, Wt_e2a, b_e2a, A, v_raw, D, D, D, VPAD);

    // 5) alpha = sigmoid(gelu([e_out|h_feat]@W_g1 + b_g1) @ W_g2 + b_g2)
    k_gate_alpha<<<NT / 128, 256, 0, stream>>>(e_out, h_feat, Wt_g1, b_g1, W_g2,
                                               b_g2, alpha, D);

    // 6) manifold chain -> z_tan (bf16, padded)
    k_manifold<<<NT, 256, 0, stream>>>(y_h, anchor, v_raw, alpha, z_tan, A, APAD,
                                       VPAD);

    // 7) z_fused = z_tan @ W_out + b_out  (f32 out -> d_out)
    k_gemm_wmma<false, false><<<dim3(D / 64, NT / 128), 256, 0, stream>>>(
        z_tan, Wt_out, b_out, D, z_fused, APAD, APAD, APAD, D);
}
